// SpikingEiNorm1d_72868415144475
// MI455X (gfx1250) — matrix operand order repaired and checked by reading, not compiled, not run on hardware
//
#include <hip/hip_runtime.h>

typedef __attribute__((ext_vector_type(16))) __bf16 v16bf;
typedef __attribute__((ext_vector_type(8)))  __bf16 v8bf;
typedef __attribute__((ext_vector_type(8)))  float  v8f;

#define BROWS 32768
#define NE    2048
#define NI    256

// ---- bf16 hi/lo split: x ~= hi + lo with |err| ~ 2^-18 |x| ----
__device__ __forceinline__ void bf16_split(float x, __bf16& hi, __bf16& lo) {
    hi = (__bf16)x;
    lo = (__bf16)(x - (float)hi);
}

// Pre-kernel: clamp weight >= 0 and pre-split into hi/lo bf16 planes (2 MB in d_ws).
__global__ void __launch_bounds__(256)
wsplit_kernel(const float* __restrict__ w, __bf16* __restrict__ whi, __bf16* __restrict__ wlo) {
    int i = blockIdx.x * 256 + threadIdx.x;   // grid sized exactly NE*NI/256
    float v = w[i];
    v = v > 0.0f ? v : 0.0f;
    __bf16 h, l;
    bf16_split(v, h, l);
    whi[i] = h;
    wlo[i] = l;
}

// A operand (16-bit 16x32): lane L: M = L&15, half h = L>>4.
// VGPR v (pair 2v,2v+1): K = (v%4)*2 + (v/4)*16 + h*8 -> lane holds K chunks
// [kb+8h .. kb+8h+7] and [kb+16+8h .. kb+16+8h+7], each 16 B contiguous in LDS.
__device__ __forceinline__ v16bf load_a(const __bf16* base, int off) {
    v16bf r;
    ((v8bf*)&r)[0] = *(const v8bf*)(base + off);
    ((v8bf*)&r)[1] = *(const v8bf*)(base + off + 16);
    return r;
}

__device__ __forceinline__ v8f wmma_bf16(v16bf a, v16bf b, v8f c) {
    // (neg_a, A, neg_b, B, c_mod, C, reuse_a, reuse_b) — probe-confirmed signature
    return __builtin_amdgcn_wmma_f32_16x16x32_bf16(false, a, false, b, (short)0, c, false, false);
}

// Dynamic LDS layout (bytes):
//   lds_num : 16*2048 f32  = 131072
//   lds_sh  : 16*2048 f32  = 131072
//   a_ie_hi/lo, a_gi_hi/lo : 4 * 16*256 bf16 = 32768
//   rowmin  : 16 f32 = 64
// total = 294976 (<= 320 KB WGP LDS)
#define SMEM_BYTES (2*16*2048*4 + 4*16*256*2 + 16*4)

__global__ void __launch_bounds__(256, 1)
spiking_einorm_kernel(const float* __restrict__ I_ee,
                      const float* __restrict__ I_ie,
                      const __bf16* __restrict__ w_hi,
                      const __bf16* __restrict__ w_lo,
                      const float* __restrict__ gain_i,
                      const float* __restrict__ gain_e,
                      const float* __restrict__ bias,
                      float* __restrict__ out) {
    extern __shared__ __align__(16) char smem_raw[];
    float*  lds_num = (float*)smem_raw;                 // [16][2048] numerator ge*(I_ee - I_ei)
    float*  lds_sh  = lds_num + 16 * NE;                // [16][2048] I_shunting
    __bf16* a_ie_hi = (__bf16*)(lds_sh + 16 * NE);      // [16][256]
    __bf16* a_ie_lo = a_ie_hi + 16 * NI;
    __bf16* a_gi_hi = a_ie_lo + 16 * NI;
    __bf16* a_gi_lo = a_gi_hi + 16 * NI;
    float*  rowmin  = (float*)(a_gi_lo + 16 * NI);      // [16]

    const int t    = threadIdx.x;
    const int row0 = blockIdx.x * 16;                   // 16-sample block

    // ---------- Stage 0: load I_ie block, build bf16-split A operands ----------
    #pragma unroll
    for (int k = 0; k < 16; ++k) {
        int e   = t + 256 * k;                          // 0..4095
        int col = e & (NI - 1);
        int r   = e >> 8;                               // 0..15
        float v = I_ie[(row0 + r) * NI + col];
        float g = gain_i[col];
        g = g > 0.0f ? g : 0.0f;
        float gv = g * v;
        __bf16 h, l;
        bf16_split(v, h, l);
        a_ie_hi[r * NI + col] = h;
        a_ie_lo[r * NI + col] = l;
        bf16_split(gv, h, l);
        a_gi_hi[r * NI + col] = h;
        a_gi_lo[r * NI + col] = l;
    }
    __syncthreads();

    const int lane   = t & 31;
    const int waveId = t >> 5;                          // 0..7, owns cols [waveId*256, +256)
    const int m      = lane & 15;                       // A: M row / B: N col
    const int h      = lane >> 4;                       // lane half

    // ---------- Stage 1: dual GEMM via bf16x3 WMMA, tiles processed in pairs ----------
    for (int tp = 0; tp < 8; ++tp) {
        const int j0 = waveId * 16 + tp * 2;            // two 16-wide column tiles j0, j0+1
        v8f cei[2] = {{}, {}};                          // I_ei accumulators
        v8f csh[2] = {{}, {}};                          // I_shunting accumulators

        #pragma unroll
        for (int kb = 0; kb < NI; kb += 32) {
            const int aoff = m * NI + kb + h * 8;
            v16bf aih = load_a(a_ie_hi, aoff);
            v16bf ail = load_a(a_ie_lo, aoff);
            v16bf agh = load_a(a_gi_hi, aoff);
            v16bf agl = load_a(a_gi_lo, aoff);

            // B (16-bit 32x16): same K order as A: chunks [kb+8h..+7] and [kb+16+8h..+7]:
            // w.T[k, n] = w[col, k], col = j*16 + m (each chunk 16 B contiguous).
            const int boff0 = (j0 * 16 + m) * NI + kb + h * 8;
            const int boff1 = boff0 + 16 * NI;          // next tile: col += 16
            v16bf b0h = load_a(w_hi, boff0);
            v16bf b0l = load_a(w_lo, boff0);
            v16bf b1h = load_a(w_hi, boff1);
            v16bf b1l = load_a(w_lo, boff1);

            // 3-term split accumulation: lo*hi, hi*lo, hi*hi
            cei[0] = wmma_bf16(ail, b0h, cei[0]);
            cei[0] = wmma_bf16(aih, b0l, cei[0]);
            cei[0] = wmma_bf16(aih, b0h, cei[0]);
            csh[0] = wmma_bf16(agl, b0h, csh[0]);
            csh[0] = wmma_bf16(agh, b0l, csh[0]);
            csh[0] = wmma_bf16(agh, b0h, csh[0]);

            cei[1] = wmma_bf16(ail, b1h, cei[1]);
            cei[1] = wmma_bf16(aih, b1l, cei[1]);
            cei[1] = wmma_bf16(aih, b1h, cei[1]);
            csh[1] = wmma_bf16(agl, b1h, csh[1]);
            csh[1] = wmma_bf16(agh, b1l, csh[1]);
            csh[1] = wmma_bf16(agh, b1h, csh[1]);
        }

        // Tile epilogue: fuse I_ee read, clamp(ge), numerator; spill tiles to LDS.
        // C/D layout: VGPR r -> (M = r + 8*h, N = lane&15).
        #pragma unroll
        for (int jj = 0; jj < 2; ++jj) {
            const int col = (j0 + jj) * 16 + m;
            float ge = gain_e[col];
            ge = ge > 0.0f ? ge : 0.0f;
            #pragma unroll
            for (int r = 0; r < 8; ++r) {
                const int mr  = h * 8 + r;
                float iee = I_ee[(size_t)(row0 + mr) * NE + col];
                lds_num[mr * NE + col] = ge * (iee - cei[jj][r]);
                lds_sh [mr * NE + col] = csh[jj][r];
            }
        }
    }
    __syncthreads();

    // ---------- Stage 2: per-row min over non-zero shunting entries ----------
    {
        const int r = t >> 4;                           // row 0..15 (16 threads/row)
        const int g = t & 15;
        float mn = __builtin_inff();
        for (int k = 0; k < NE / 16; ++k) {
            float v = lds_sh[r * NE + g + 16 * k];
            v = (v == 0.0f) ? __builtin_inff() : v;
            mn = v < mn ? v : mn;
        }
        #pragma unroll
        for (int mask = 1; mask <= 8; mask <<= 1) {
            float o = __shfl_xor(mn, mask, 32);         // 16-lane group stays in-wave
            mn = o < mn ? o : mn;
        }
        if (g == 0) rowmin[r] = mn;
    }
    __syncthreads();

    // ---------- Stage 3: finalize and store (coalesced) ----------
    for (int k = 0; k < 16 * NE / 256; ++k) {
        const int e   = t + 256 * k;
        const int r   = e >> 11;
        const int col = e & (NE - 1);
        float sh  = lds_sh[r * NE + col];
        float adj = (sh == 0.0f) ? rowmin[r] : sh;
        out[(size_t)(row0 + r) * NE + col] = lds_num[r * NE + col] / adj + bias[col];
    }
}

extern "C" void kernel_launch(void* const* d_in, const int* in_sizes, int n_in,
                              void* d_out, int out_size, void* d_ws, size_t ws_size,
                              hipStream_t stream) {
    const float* I_ee = (const float*)d_in[0];
    const float* I_ie = (const float*)d_in[1];
    const float* w    = (const float*)d_in[2];
    const float* gi   = (const float*)d_in[3];
    const float* ge   = (const float*)d_in[4];
    const float* bias = (const float*)d_in[5];
    float* out = (float*)d_out;

    __bf16* w_hi = (__bf16*)d_ws;                        // NE*NI bf16 (1 MB)
    __bf16* w_lo = w_hi + NE * NI;                       // NE*NI bf16 (1 MB)

    // Pre-split clamped weights into bf16 hi/lo planes (L2-resident, reused by all blocks).
    wsplit_kernel<<<(NE * NI) / 256, 256, 0, stream>>>(w, w_hi, w_lo);

    // 289 KB dynamic LDS per workgroup (gfx1250 WGP: 320 KB).
    hipFuncSetAttribute((const void*)spiking_einorm_kernel,
                        hipFuncAttributeMaxDynamicSharedMemorySize, SMEM_BYTES);

    spiking_einorm_kernel<<<BROWS / 16, 256, SMEM_BYTES, stream>>>(
        I_ee, I_ie, w_hi, w_lo, gi, ge, bias, out);
}
